// GraphAttentionLayer_36069135352324
// MI455X (gfx1250) — matrix operand order repaired and checked by reading, not compiled, not run on hardware
//
#include <hip/hip_runtime.h>
#include <hip/hip_bf16.h>
#include <math.h>

#define B_   4
#define N_   512
#define DIN  128
#define DOUT 64

typedef __attribute__((ext_vector_type(16))) __bf16 v16bf;
typedef __attribute__((ext_vector_type(8)))  __bf16 v8bf;
typedef __attribute__((ext_vector_type(8)))  float  v8f;

// A-matrix (16x32 bf16) fragment layout per CDNA5 ISA:
// lanes 0-15  (half=0): K = d0..d0+7 in elems 0..7, K = d0+16..d0+23 in elems 8..15
// lanes 16-31 (half=1): same pattern with d0 already offset by +8 by the caller.
__device__ __forceinline__ v16bf load_fragA(const __bf16* row, int d0) {
  v8bf lo = *reinterpret_cast<const v8bf*>(row + d0);
  v8bf hi = *reinterpret_cast<const v8bf*>(row + d0 + 16);
  v16bf f;
#pragma unroll
  for (int t = 0; t < 8; ++t) { f[t] = lo[t]; f[t + 8] = hi[t]; }
  return f;
}

// ---------------- kernel 0: x -> bf16 ----------------
__global__ void k_tobf16(const float* __restrict__ x, __bf16* __restrict__ xb) {
  int t = blockIdx.x * 256 + threadIdx.x;   // B*N*DIN = 262144
  xb[t] = (__bf16)x[t];
}

// ---------------- kernel 1: bilinear tanh scores via WMMA ----------------
// scores[b,i,j] = sum_o v_o * tanh( (x_i .* w_o) . x_j + b_o )
// Single af buffer: keeps allocation at ~244 VGPRs (no s_set_vgpr_msb churn)
// and lets the scalar bf16 multiplies pattern-match to v_pk_mul_bf16. The 4
// co-exec hazard NOPs per 4-WMMA group largely overlap the XDL pipe.
__global__ __launch_bounds__(128, 1)
void k_scores(const __bf16* __restrict__ Xbf,
              const float* __restrict__ att_w,   // (64,128)
              const float* __restrict__ att_b,   // (64)
              const float* __restrict__ att_v,   // (64,1) -> 64
              float* __restrict__ scores) {
  __shared__ __bf16 wlds[DOUT * DIN];            // 16 KB
  const int tid = threadIdx.x;
  for (int k = tid; k < DOUT * DIN; k += 128) wlds[k] = (__bf16)att_w[k];
  __syncthreads();

  const int wave = tid >> 5, lane = tid & 31;
  const int m = lane & 15, half = lane >> 4;

  const int g  = blockIdx.x * 4 + wave;          // 1024 wave-tiles
  const int b  = g >> 8;                         // 0..3
  const int it = (g >> 3) & 31;                  // i-tile (16 rows)
  const int jc = g & 7;                          // j-chunk (64 cols)
  const int ibase = it * 16, jbase = jc * 64;

  const __bf16* Xb = Xbf + (size_t)b * N_ * DIN;

  // Persistent B fragments: Xt (128 x 16) per j-subtile, per 32-K chunk.
  v16bf Bf[4][4];
#pragma unroll
  for (int jt = 0; jt < 4; ++jt) {
    const __bf16* row = Xb + (size_t)(jbase + jt * 16 + m) * DIN;
#pragma unroll
    for (int kc = 0; kc < 4; ++kc)
      Bf[jt][kc] = load_fragA(row, kc * 32 + half * 8);
  }

  // Persistent x_i fragments in A layout (rescaled by w_o each o-iteration).
  v16bf Xi[4];
  {
    const __bf16* row = Xb + (size_t)(ibase + m) * DIN;
#pragma unroll
    for (int kc = 0; kc < 4; ++kc)
      Xi[kc] = load_fragA(row, kc * 32 + half * 8);
  }

  v8f S[4];
#pragma unroll
  for (int jt = 0; jt < 4; ++jt)
#pragma unroll
    for (int r = 0; r < 8; ++r) S[jt][r] = 0.f;

  for (int o = 0; o < DOUT; ++o) {
    const float vo = att_v[o];
    const float bo = att_b[o];
    const __bf16* wrow = &wlds[o * DIN];

    v8f acc[4];
#pragma unroll
    for (int jt = 0; jt < 4; ++jt)
#pragma unroll
      for (int r = 0; r < 8; ++r) acc[jt][r] = 0.f;

#pragma unroll
    for (int kc = 0; kc < 4; ++kc) {
      v16bf wf = load_fragA(wrow, kc * 32 + half * 8);   // ds_load
      v16bf af;
#pragma unroll
      for (int t = 0; t < 16; ++t)
        af[t] = (__bf16)((float)Xi[kc][t] * (float)wf[t]);
#pragma unroll
      for (int jt = 0; jt < 4; ++jt)
        acc[jt] = __builtin_amdgcn_wmma_f32_16x16x32_bf16(
            false, af, false, Bf[jt][kc], (short)0, acc[jt], false, false);
    }

#pragma unroll
    for (int jt = 0; jt < 4; ++jt)
#pragma unroll
      for (int r = 0; r < 8; ++r)
        S[jt][r] += vo * tanhf(acc[jt][r] + bo);
  }

  // C/D layout: element r of lane -> row = r + 8*half, col = m (per 16x16 f32 table)
  const int irow = ibase + half * 8;
#pragma unroll
  for (int jt = 0; jt < 4; ++jt) {
    const int col = jbase + jt * 16 + m;
#pragma unroll
    for (int r = 0; r < 8; ++r)
      scores[(size_t)(b * N_ + irow + r) * N_ + col] = S[jt][r];
  }
}

// ---------------- kernel 2: softmax over j (one wave per row) ----------------
__global__ __launch_bounds__(256)
void k_softmax(float* __restrict__ scores) {
  const int tid  = threadIdx.x;
  const int row  = blockIdx.x * 8 + (tid >> 5);   // 2048 rows
  const int lane = tid & 31;
  float* s = scores + (size_t)row * N_;

  float v[16];
  float mx = -1e30f;
#pragma unroll
  for (int t = 0; t < 16; ++t) { v[t] = s[lane + t * 32]; mx = fmaxf(mx, v[t]); }
#pragma unroll
  for (int d = 16; d; d >>= 1) mx = fmaxf(mx, __shfl_xor(mx, d, 32));
  float sum = 0.f;
#pragma unroll
  for (int t = 0; t < 16; ++t) { v[t] = __expf(v[t] - mx); sum += v[t]; }
#pragma unroll
  for (int d = 16; d; d >>= 1) sum += __shfl_xor(sum, d, 32);
  const float inv = 1.0f / sum;
#pragma unroll
  for (int t = 0; t < 16; ++t) s[lane + t * 32] = v[t] * inv;
}

// ---------------- kernel 3: agg = att @ x (f32) ----------------
__global__ __launch_bounds__(256)
void k_agg(const float* __restrict__ att, const float* __restrict__ x,
           float* __restrict__ agg) {
  const int t = blockIdx.x * 256 + threadIdx.x;   // B*N*DIN
  const int d = t & (DIN - 1);
  const int i = (t >> 7) & (N_ - 1);
  const int b = t >> 16;
  const float* arow = att + (size_t)(b * N_ + i) * N_;
  const float* xb   = x + (size_t)b * N_ * DIN + d;
  float a = 0.f;
  for (int j = 0; j < N_; ++j) a += arow[j] * xb[(size_t)j * DIN];
  agg[t] = a;
}

// ---------------- kernel 4: h = agg@W1^T + b1 + x@W2^T + b2 ----------------
__global__ __launch_bounds__(256)
void k_proj(const float* __restrict__ agg, const float* __restrict__ x,
            const float* __restrict__ w1, const float* __restrict__ b1,
            const float* __restrict__ w2, const float* __restrict__ b2,
            float* __restrict__ h) {
  const int t  = blockIdx.x * 256 + threadIdx.x;  // B*N*DOUT = 131072
  const int o  = t & (DOUT - 1);
  const int bi = t >> 6;                          // 0..2047
  const float* ar  = agg + (size_t)bi * DIN;
  const float* xr  = x   + (size_t)bi * DIN;
  const float* w1r = w1  + (size_t)o * DIN;
  const float* w2r = w2  + (size_t)o * DIN;
  float a = b1[o] + b2[o];
  for (int d = 0; d < DIN; ++d) a += ar[d] * w1r[d] + xr[d] * w2r[d];
  h[t] = a;
}

// ---------------- kernel 5: batch-stat BN + SELU (one block per channel) ----
__global__ __launch_bounds__(256)
void k_bn_selu(const float* __restrict__ h, const float* __restrict__ gamma,
               const float* __restrict__ beta, float* __restrict__ out) {
  __shared__ float s1[256], s2[256];
  const int o = blockIdx.x, tid = threadIdx.x;
  const int R = B_ * N_;                          // 2048
  float sum = 0.f, sq = 0.f;
  for (int r = tid; r < R; r += 256) {
    float v = h[(size_t)r * DOUT + o];
    sum += v; sq += v * v;
  }
  s1[tid] = sum; s2[tid] = sq;
  __syncthreads();
  for (int st = 128; st; st >>= 1) {
    if (tid < st) { s1[tid] += s1[tid + st]; s2[tid] += s2[tid + st]; }
    __syncthreads();
  }
  const float mean = s1[0] / (float)R;
  const float var  = s2[0] / (float)R - mean * mean;
  const float inv  = rsqrtf(var + 1e-5f) * gamma[o];
  const float bt   = beta[o];
  const float scale = 1.0507009873554805f, alpha = 1.6732632423543772f;
  for (int r = tid; r < R; r += 256) {
    float v = (h[(size_t)r * DOUT + o] - mean) * inv + bt;
    out[(size_t)r * DOUT + o] = v > 0.f ? scale * v : scale * alpha * (__expf(v) - 1.f);
  }
}

extern "C" void kernel_launch(void* const* d_in, const int* in_sizes, int n_in,
                              void* d_out, int out_size, void* d_ws, size_t ws_size,
                              hipStream_t stream) {
  const float* x      = (const float*)d_in[0];
  const float* att_w  = (const float*)d_in[1];
  const float* att_b  = (const float*)d_in[2];
  const float* att_v  = (const float*)d_in[3];
  const float* w1     = (const float*)d_in[4];
  const float* b1     = (const float*)d_in[5];
  const float* w2     = (const float*)d_in[6];
  const float* b2     = (const float*)d_in[7];
  const float* gamma  = (const float*)d_in[8];
  const float* beta   = (const float*)d_in[9];
  float* out = (float*)d_out;

  char* ws = (char*)d_ws;
  __bf16* Xbf   = (__bf16*)ws;                                   // 512 KB
  float*  score = (float*)(ws + (512 << 10));                    // 4 MB
  float*  agg   = (float*)(ws + (512 << 10) + (4 << 20));        // 1 MB
  float*  h     = (float*)(ws + (512 << 10) + (5 << 20));        // 512 KB

  k_tobf16 <<<(B_ * N_ * DIN) / 256, 256, 0, stream>>>(x, Xbf);
  k_scores <<<256, 128, 0, stream>>>(Xbf, att_w, att_b, att_v, score);
  k_softmax<<<256, 256, 0, stream>>>(score);
  k_agg    <<<(B_ * N_ * DIN) / 256, 256, 0, stream>>>(score, x, agg);
  k_proj   <<<(B_ * N_ * DOUT) / 256, 256, 0, stream>>>(agg, x, w1, b1, w2, b2, h);
  k_bn_selu<<<DOUT, 256, 0, stream>>>(h, gamma, beta, out);
}
